// BiLSTMCRF_27590869910194
// MI455X (gfx1250) — matrix operand order repaired and checked by reading, not compiled, not run on hardware
//
#include <hip/hip_runtime.h>
#include <hip/hip_bf16.h>
#include <math.h>

// ---------------- problem constants ----------------
#define V_  100000
#define K_  32
#define E_  128
#define H_  128
#define B_  64
#define T_  1024
#define G4H 512   // 4*H

#define XSTRIDE 136   // padded half-stride for x/h LDS tiles (breaks bank aliasing)
#define GSTRIDE 17    // padded float-stride for gate LDS buffer

typedef __attribute__((ext_vector_type(16))) _Float16 v16h;
typedef __attribute__((ext_vector_type(8)))  float    v8f;

// ---------------- WMMA operand builders (CDNA5 16x16x32 f16 layouts) ----------
// A operand: 16x32 f16 tile, rows = M. Lane l<16: elements 0..15 = K k0..k0+15;
// lane>=16: K k0+16..k0+31.
__device__ inline v16h make_A(const _Float16* __restrict__ W, int rowBase,
                              int rowStride, int k0, int lane) {
  const _Float16* r = W + (size_t)(rowBase + (lane & 15)) * rowStride;
  const int base = (lane >= 16) ? 16 : 0;
  v16h a;
#pragma unroll
  for (int e = 0; e < 8; ++e) {
    a[e]     = r[k0 + base + e];
    a[e + 8] = r[k0 + base + 8 + e];
  }
  return a;
}

// B operand: 32x16 f16 tile, cols = N (lane&15). Lanes 0-15 hold K k0..k0+15,
// lanes 16-31 hold K k0+16..k0+31 (contiguous k order, same as make_A).
__device__ inline v16h make_B(const _Float16* __restrict__ colBase,
                              int colStride, int k0, int lane) {
  const _Float16* c = colBase + (size_t)(lane & 15) * colStride;
  const int base = (lane >= 16) ? 16 : 0;
  v16h b;
#pragma unroll
  for (int e = 0; e < 16; ++e) b[e] = c[k0 + base + e];
  return b;
}

__device__ inline v8f zero8() {
  v8f z;
#pragma unroll
  for (int i = 0; i < 8; ++i) z[i] = 0.0f;
  return z;
}

__device__ inline float fast_sigmoid(float x) {
  return 1.0f / (1.0f + __expf(-x));
}
__device__ inline float fast_tanh(float x) {
  float e = __expf(2.0f * x);
  return 1.0f - 2.0f / (e + 1.0f);
}

// ---------------- prep kernels ----------------
// Wcat[row][0..255] = concat(w_ih[row][0..127], w_hh[row][0..127]) as f16
__global__ void prep_wcat(const float* __restrict__ w_ih,
                          const float* __restrict__ w_hh,
                          _Float16* __restrict__ wcat) {
  int i = blockIdx.x * blockDim.x + threadIdx.x;
  if (i >= G4H * 256) return;
  int row = i >> 8, k = i & 255;
  float v = (k < H_) ? w_ih[row * E_ + k] : w_hh[row * H_ + (k - H_)];
  wcat[i] = (_Float16)v;
}

__global__ void prep_bias(const float* __restrict__ bi,
                          const float* __restrict__ bh,
                          float* __restrict__ bias) {
  int i = blockIdx.x * blockDim.x + threadIdx.x;
  if (i < G4H) bias[i] = bi[i] + bh[i];
}

__global__ void prep_fcw(const float* __restrict__ fcw,
                         _Float16* __restrict__ out) {
  int i = blockIdx.x * blockDim.x + threadIdx.x;
  if (i < K_ * 2 * H_) out[i] = (_Float16)fcw[i];
}

// x_f16[t][b][e] = emb[tokens[b][t]][e]   (t-major for the recurrence)
__global__ void embed_kernel(const int* __restrict__ tokens,
                             const float* __restrict__ emb,
                             _Float16* __restrict__ x) {
  size_t i = (size_t)blockIdx.x * blockDim.x + threadIdx.x;
  if (i >= (size_t)T_ * B_ * E_) return;
  int e = (int)(i & (E_ - 1));
  size_t tb = i >> 7;                 // t*B + b
  int b = (int)(tb & (B_ - 1));
  int t = (int)(tb >> 6);
  int tok = tokens[(size_t)b * T_ + t];
  x[i] = (_Float16)emb[(size_t)tok * E_ + e];
}

// ---------------- persistent BiLSTM recurrence ----------------
// grid (4, 2): blockIdx.x = batch group (16 rows), blockIdx.y = direction.
// Computes G^T = Wcat(512x256) x [x;h]^T(256x16) with weights VGPR-resident.
__global__ __launch_bounds__(256) void lstm_kernel(
    const _Float16* __restrict__ x,       // [T][B][E] f16
    const _Float16* __restrict__ wcatF,   // [512][256] f16
    const _Float16* __restrict__ wcatB,
    const float* __restrict__ biasF,      // [512]
    const float* __restrict__ biasB,
    _Float16* __restrict__ hF,            // [T][B][H] f16
    _Float16* __restrict__ hB) {
  const int tid  = threadIdx.x;
  const int lane = tid & 31;
  const int wave = tid >> 5;              // 0..7, owns gate rows [64w, 64w+64)
  const int bg   = blockIdx.x;            // batch group 0..3
  const int dir  = blockIdx.y;            // 0 fwd, 1 bwd

  const _Float16* Wc  = dir ? wcatB : wcatF;
  const float*    bia = dir ? biasB : biasF;
  _Float16*       ho  = dir ? hB : hF;

  __shared__ __align__(16) _Float16 x_lds[16 * XSTRIDE]; // staged x tile
  __shared__ __align__(16) _Float16 h_lds[16 * XSTRIDE]; // current h (f16)
  __shared__ float    c_lds[16 * H_];                    // cell state
  __shared__ float    G[G4H * GSTRIDE];                  // gates [row][batch]
  __shared__ float    bias_lds[G4H];

  for (int i = tid; i < G4H; i += 256) bias_lds[i] = bia[i];
  for (int i = tid; i < 16 * XSTRIDE; i += 256) h_lds[i] = (_Float16)0.0f;
  for (int i = tid; i < 16 * H_; i += 256) c_lds[i] = 0.0f;

  // Persistent A operands: this wave's 64 gate rows, full K=256.
  // 4 M-tiles x 8 K-tiles x 8 VGPRs = 256 VGPRs of weights per lane.
  v16h Areg[4][8];
#pragma unroll
  for (int mt = 0; mt < 4; ++mt)
#pragma unroll
    for (int kt = 0; kt < 8; ++kt)
      Areg[mt][kt] = make_A(Wc, 64 * wave + 16 * mt, 256, 32 * kt, lane);

  // x staging: 128 threads each own one 16B chunk (row = tid>>3, seg = tid&7)
  const bool stager = tid < 128;
  const int  srow = tid >> 3;
  const int  sseg = tid & 7;
  const size_t gstep = (size_t)B_ * E_;   // halves per timestep
  const _Float16* xbase =
      x + ((size_t)(bg * 16) + srow) * E_ + sseg * 8;   // + t*gstep
  _Float16* ldst = &x_lds[srow * XSTRIDE + sseg * 8];

  {
    const int t0 = dir ? (T_ - 1) : 0;
    if (stager) *(uint4*)ldst = *(const uint4*)(xbase + (size_t)t0 * gstep);
  }
  __syncthreads();

  for (int s = 0; s < T_; ++s) {
    const int t = dir ? (T_ - 1 - s) : s;

    // issue next tile's global load now; latency hides behind WMMA+elementwise
    uint4 xnext;
    if (stager && (s + 1 < T_)) {
      const int tn = dir ? (T_ - 2 - s) : (s + 1);
      xnext = *(const uint4*)(xbase + (size_t)tn * gstep);
      if (s + 4 < T_) {
        const int tp = dir ? (T_ - 5 - s) : (s + 4);
        __builtin_prefetch(xbase + (size_t)tp * gstep, 0, 1);
      }
    }

    v8f acc[4];
#pragma unroll
    for (int mt = 0; mt < 4; ++mt) acc[mt] = zero8();

#pragma unroll
    for (int kt = 0; kt < 8; ++kt) {
      v16h b = (kt < 4) ? make_B(x_lds, XSTRIDE, 32 * kt, lane)
                        : make_B(h_lds, XSTRIDE, 32 * (kt - 4), lane);
#pragma unroll
      for (int mt = 0; mt < 4; ++mt)
        acc[mt] = __builtin_amdgcn_wmma_f32_16x16x32_f16(
            false, Areg[mt][kt], false, b, (short)0, acc[mt], false, false);
    }

    // D layout: lane gives batch col (N = lane&15); VGPR r -> gate row r (+8)
    {
      const int col  = lane & 15;
      const int radd = (lane >= 16) ? 8 : 0;
#pragma unroll
      for (int mt = 0; mt < 4; ++mt) {
        const int row0 = 64 * wave + 16 * mt + radd;
#pragma unroll
        for (int r = 0; r < 8; ++r) G[(row0 + r) * GSTRIDE + col] = acc[mt][r];
      }
    }
    __syncthreads();

    // elementwise LSTM cell: 16x128 elems, 8 per thread
#pragma unroll
    for (int u = 0; u < 8; ++u) {
      const int idx = tid + u * 256;   // = b_local*128 + j
      const int bl  = idx >> 7;
      const int j   = idx & 127;
      float gi = G[(j)*GSTRIDE + bl]            + bias_lds[j];
      float gf = G[(H_ + j) * GSTRIDE + bl]     + bias_lds[H_ + j];
      float gg = G[(2 * H_ + j) * GSTRIDE + bl] + bias_lds[2 * H_ + j];
      float go = G[(3 * H_ + j) * GSTRIDE + bl] + bias_lds[3 * H_ + j];
      float i_ = fast_sigmoid(gi);
      float f_ = fast_sigmoid(gf);
      float o_ = fast_sigmoid(go);
      float c_ = f_ * c_lds[idx] + i_ * fast_tanh(gg);
      float h_ = o_ * fast_tanh(c_);
      c_lds[idx] = c_;
      _Float16 hh = (_Float16)h_;
      h_lds[bl * XSTRIDE + j] = hh;
      ho[((size_t)t * B_ + bg * 16 + bl) * H_ + j] = hh;
    }

    // stage next x tile (x_lds not read again until after the barrier)
    if (stager && (s + 1 < T_)) *(uint4*)ldst = xnext;
    __syncthreads();
  }
}

// ---------------- emissions: em = [hf|hb] @ fc_w^T + fc_b ----------------
// 8192 tiles (4096 M-tiles x 2 N-tiles), one wave per tile, K = 256.
__global__ __launch_bounds__(128) void emis_kernel(
    const _Float16* __restrict__ hf, const _Float16* __restrict__ hb,
    const _Float16* __restrict__ fcw,   // [32][256] f16
    const float* __restrict__ fcb, float* __restrict__ em) {
  const int lane = threadIdx.x & 31;
  const int tile = blockIdx.x * 4 + (threadIdx.x >> 5);   // 0..8191
  const int nt = tile & 1;
  const int mt = tile >> 1;
  const int row0 = mt * 16;

  v8f acc = zero8();
#pragma unroll
  for (int kt = 0; kt < 8; ++kt) {
    const _Float16* src = (kt < 4) ? hf : hb;
    const int k0 = (kt < 4) ? 32 * kt : 32 * (kt - 4);
    v16h a = make_A(src, row0, H_, k0, lane);
    v16h b = make_B(fcw, 256, 32 * kt, lane);
    acc = __builtin_amdgcn_wmma_f32_16x16x32_f16(false, a, false, b, (short)0,
                                                 acc, false, false);
  }
  const int n    = nt * 16 + (lane & 15);
  const float fb = fcb[n];
  const int radd = (lane >= 16) ? 8 : 0;
#pragma unroll
  for (int r = 0; r < 8; ++r)
    em[(size_t)(row0 + r + radd) * K_ + n] = acc[r] + fb;
}

// ---------------- CRF path score ----------------
__global__ void crf_score_kernel(const float* __restrict__ em,
                                 const int* __restrict__ tags,      // [B][T]
                                 const unsigned char* __restrict__ mask,
                                 const float* __restrict__ start_t,
                                 const float* __restrict__ end_t,
                                 const float* __restrict__ trans,
                                 float* __restrict__ score) {
  int b = threadIdx.x;
  if (b >= B_) return;
  int tg0 = tags[(size_t)b * T_];
  float sc = start_t[tg0] + em[(size_t)b * K_ + tg0];
  int prev = tg0;
  int msum = mask[(size_t)b * T_] ? 1 : 0;
  for (int t = 1; t < T_; ++t) {
    int tg = tags[(size_t)b * T_ + t];
    if (mask[(size_t)b * T_ + t]) {
      sc += trans[prev * K_ + tg] + em[((size_t)t * B_ + b) * K_ + tg];
      msum++;
    }
    prev = tg;
  }
  int last_tag = tags[(size_t)b * T_ + (msum - 1)];
  score[b] = sc + end_t[last_tag];
}

// ---------------- CRF forward (alpha) recursion ----------------
__global__ void crf_alpha_kernel(const float* __restrict__ em,
                                 const unsigned char* __restrict__ mask,
                                 const float* __restrict__ start_t,
                                 const float* __restrict__ end_t,
                                 const float* __restrict__ trans,
                                 float* __restrict__ denom) {
  const int b = blockIdx.x;     // batch row
  const int k = threadIdx.x;    // target tag, 32 threads
  __shared__ float alpha[K_];
  __shared__ float tr[K_ * K_];
  __shared__ float red[K_];
  for (int i = threadIdx.x; i < K_ * K_; i += K_) tr[i] = trans[i];
  alpha[k] = start_t[k] + em[(size_t)b * K_ + k];
  __syncthreads();
  for (int t = 1; t < T_; ++t) {
    float m = -1e30f;
#pragma unroll
    for (int kk = 0; kk < K_; ++kk) m = fmaxf(m, alpha[kk] + tr[kk * K_ + k]);
    float s = 0.0f;
#pragma unroll
    for (int kk = 0; kk < K_; ++kk) s += __expf(alpha[kk] + tr[kk * K_ + k] - m);
    float nxt = m + __logf(s) + em[((size_t)t * B_ + b) * K_ + k];
    unsigned char mt = mask[(size_t)b * T_ + t];
    __syncthreads();
    if (mt) alpha[k] = nxt;
    __syncthreads();
  }
  red[k] = alpha[k] + end_t[k];
  __syncthreads();
  if (k == 0) {
    float m = -1e30f;
    for (int i = 0; i < K_; ++i) m = fmaxf(m, red[i]);
    float s = 0.0f;
    for (int i = 0; i < K_; ++i) s += __expf(red[i] - m);
    denom[b] = m + __logf(s);
  }
}

// ---------------- finalize: loss = -mean(score - denom) ----------------
__global__ void final_kernel(const float* __restrict__ score,
                             const float* __restrict__ denom,
                             float* __restrict__ out) {
  __shared__ float red[B_];
  int i = threadIdx.x;
  if (i < B_) red[i] = score[i] - denom[i];
  __syncthreads();
  if (i == 0) {
    float s = 0.0f;
    for (int j = 0; j < B_; ++j) s += red[j];
    out[0] = -s / (float)B_;
  }
}

// ---------------- host launcher ----------------
extern "C" void kernel_launch(void* const* d_in, const int* in_sizes, int n_in,
                              void* d_out, int out_size, void* d_ws,
                              size_t ws_size, hipStream_t stream) {
  const int*   tokens = (const int*)d_in[0];
  const int*   tags   = (const int*)d_in[1];
  const unsigned char* mask = (const unsigned char*)d_in[2];
  const float* emb    = (const float*)d_in[3];
  const float* w_ih_f = (const float*)d_in[4];
  const float* w_hh_f = (const float*)d_in[5];
  const float* b_ih_f = (const float*)d_in[6];
  const float* b_hh_f = (const float*)d_in[7];
  const float* w_ih_b = (const float*)d_in[8];
  const float* w_hh_b = (const float*)d_in[9];
  const float* b_ih_b = (const float*)d_in[10];
  const float* b_hh_b = (const float*)d_in[11];
  const float* fc_w   = (const float*)d_in[12];
  const float* fc_b   = (const float*)d_in[13];
  const float* start_t= (const float*)d_in[14];
  const float* end_t  = (const float*)d_in[15];
  const float* trans  = (const float*)d_in[16];

  // ---- workspace carve (all offsets 256B aligned) ----
  char* w = (char*)d_ws;
  _Float16* x16   = (_Float16*)w;  w += (size_t)T_ * B_ * E_ * 2;   // 16 MB
  _Float16* hf    = (_Float16*)w;  w += (size_t)T_ * B_ * H_ * 2;   // 16 MB
  _Float16* hb    = (_Float16*)w;  w += (size_t)T_ * B_ * H_ * 2;   // 16 MB
  _Float16* wcatF = (_Float16*)w;  w += (size_t)G4H * 256 * 2;
  _Float16* wcatB = (_Float16*)w;  w += (size_t)G4H * 256 * 2;
  _Float16* fcw16 = (_Float16*)w;  w += (size_t)K_ * 2 * H_ * 2;
  float*    biasF = (float*)w;     w += (size_t)G4H * 4;
  float*    biasB = (float*)w;     w += (size_t)G4H * 4;
  float*    em    = (float*)w;     w += (size_t)T_ * B_ * K_ * 4;   // 8 MB
  float*    score = (float*)w;     w += 256;
  float*    denom = (float*)w;     w += 256;

  // ---- prep ----
  prep_wcat<<<G4H, 256, 0, stream>>>(w_ih_f, w_hh_f, wcatF);
  prep_wcat<<<G4H, 256, 0, stream>>>(w_ih_b, w_hh_b, wcatB);
  prep_bias<<<2, 256, 0, stream>>>(b_ih_f, b_hh_f, biasF);
  prep_bias<<<2, 256, 0, stream>>>(b_ih_b, b_hh_b, biasB);
  prep_fcw<<<32, 256, 0, stream>>>(fc_w, fcw16);
  embed_kernel<<<(T_ * B_ * E_) / 256, 256, 0, stream>>>(tokens, emb, x16);

  // ---- BiLSTM (both directions concurrent, 4 batch-groups each) ----
  lstm_kernel<<<dim3(4, 2), 256, 0, stream>>>(x16, wcatF, wcatB, biasF, biasB,
                                              hf, hb);

  // ---- emissions ----
  emis_kernel<<<2048, 128, 0, stream>>>(hf, hb, fcw16, fc_b, em);

  // ---- CRF ----
  crf_score_kernel<<<1, 64, 0, stream>>>(em, tags, mask, start_t, end_t, trans,
                                         score);
  crf_alpha_kernel<<<B_, K_, 0, stream>>>(em, mask, start_t, end_t, trans,
                                          denom);
  final_kernel<<<1, 64, 0, stream>>>(score, denom, (float*)d_out);
}
